// OpponnetModelling_43344809951450
// MI455X (gfx1250) — matrix operand order repaired and checked by reading, not compiled, not run on hardware
//
#include <hip/hip_runtime.h>
#include <hip/hip_bf16.h>
#include <math.h>
#include <stdint.h>

typedef __attribute__((ext_vector_type(16))) __bf16 v16bf;
typedef __attribute__((ext_vector_type(8)))  __bf16 v8bf;
typedef __attribute__((ext_vector_type(8)))  float  v8f;

// ---- problem dims ----
#define BSZ   4096
#define INF   2056
#define INP   2080   // K padded to mult of 32
#define MFP   32     // m_f (24) padded
#define NHEAD 4
#define EDIM  64
#define RNN   256
#define EN_START 4
#define AL_START 1028
#define OTH_START 2036
#define KMIX  8192   // E*NA

// =============================================================
// CDNA5 async copy helpers (ASYNCcnt path)
// =============================================================
__device__ __forceinline__ void async_copy16(void* lds_dst, const void* gsrc) {
  uint32_t l = (uint32_t)(uintptr_t)lds_dst;      // low 32 bits = LDS offset
  uint64_t g = (uint64_t)(uintptr_t)gsrc;
  asm volatile("global_load_async_to_lds_b128 %0, %1, off"
               :: "v"(l), "v"(g) : "memory");
}
__device__ __forceinline__ void async_wait_le0() {
  asm volatile("s_wait_asynccnt 0x0" ::: "memory");
}
__device__ __forceinline__ void async_wait_le3() {
  asm volatile("s_wait_asynccnt 0x3" ::: "memory");
}

__device__ __forceinline__ v16bf cat16(v8bf lo, v8bf hi) {
  return __builtin_shufflevector(lo, hi, 0, 1, 2, 3, 4, 5, 6, 7,
                                 8, 9, 10, 11, 12, 13, 14, 15);
}

// =============================================================
// conversion / packing kernels (f32 -> bf16, swizzled B layouts)
// B swizzle: idx = (((ktile*nBlk + nblk)*4 + jj)*32 + lane)*16 + k2
//   k = ktile*32 + (lane>=16 ? 16 : 0) + k2 ;  n = nblk*64 + jj*16 + (lane&15)
// =============================================================
__global__ __launch_bounds__(256) void k_cvt_obs(const float* __restrict__ obs,
                                                 __bf16* __restrict__ dst) {
  long tot = (long)BSZ * INP;
  for (long i = (long)blockIdx.x * blockDim.x + threadIdx.x; i < tot;
       i += (long)gridDim.x * blockDim.x) {
    long b = i / INP; int c = (int)(i % INP);
    float v = (c < INF) ? obs[b * (long)INF + c] : 0.f;
    dst[i] = (__bf16)v;
  }
}

__global__ __launch_bounds__(256) void k_cvt_m(const float* __restrict__ obs,
                                               __bf16* __restrict__ dst) {
  long tot = (long)BSZ * MFP;
  for (long i = (long)blockIdx.x * blockDim.x + threadIdx.x; i < tot;
       i += (long)gridDim.x * blockDim.x) {
    long b = i / MFP; int c = (int)(i % MFP);
    float v = 0.f;
    if (c < 4) v = obs[b * (long)INF + c];
    else if (c < 24) v = obs[b * (long)INF + OTH_START + (c - 4)];
    dst[i] = (__bf16)v;
  }
}

__device__ __forceinline__ void bswz(long idx, int nBlk, int& k, int& n) {
  int k2 = (int)(idx & 15);
  int lane = (int)((idx >> 4) & 31);
  int jj = (int)((idx >> 9) & 3);
  long rest = idx >> 11;
  int nblk = (int)(rest % nBlk);
  int ktile = (int)(rest / nBlk);
  k = ktile * 32 + ((lane & 16) ? 16 : 0) + k2;
  n = nblk * 64 + jj * 16 + (lane & 15);
}

// W1big^T swizzled: K=2080, N=192 (nBlk=3); also bias192
__global__ __launch_bounds__(256) void k_cvt_w1big(const float* __restrict__ Wb1,
                                                   const float* __restrict__ Wh1,
                                                   const float* __restrict__ bb1,
                                                   const float* __restrict__ bh1,
                                                   __bf16* __restrict__ dst,
                                                   float* __restrict__ bias) {
  long tot = (long)INP * 192;
  for (long i = (long)blockIdx.x * blockDim.x + threadIdx.x; i < tot;
       i += (long)gridDim.x * blockDim.x) {
    int k, n; bswz(i, 3, k, n);
    float v = 0.f;
    if (k < INF) v = (n < 64) ? Wb1[(long)n * INF + k] : Wh1[(long)(n - 64) * INF + k];
    dst[i] = (__bf16)v;
    if (i < 192) bias[i] = (i < 64) ? bb1[i] : bh1[i - 64];
  }
}

// W1stack^T swizzled: K=32, N=1536 (nBlk=24), col s = t*512 + h*128 + g
__global__ __launch_bounds__(256) void k_cvt_w1stack(const float* __restrict__ Wq1,
                                                     const float* __restrict__ Wkm1,
                                                     const float* __restrict__ Wvm1,
                                                     const float* __restrict__ bq1,
                                                     const float* __restrict__ bkm1,
                                                     const float* __restrict__ bvm1,
                                                     __bf16* __restrict__ dst,
                                                     float* __restrict__ bias) {
  long tot = (long)MFP * 1536;
  for (long i = (long)blockIdx.x * blockDim.x + threadIdx.x; i < tot;
       i += (long)gridDim.x * blockDim.x) {
    int k, s; bswz(i, 24, k, s);
    int t = s / 512, h = (s % 512) / 128, g = s % 128;
    const float* W = (t == 0) ? Wq1 : (t == 1) ? Wkm1 : Wvm1;
    dst[i] = (__bf16)((k < 24) ? W[h * 3072 + g * 24 + k] : 0.f);
    if (i < 1536) {
      int s2 = (int)i;
      int t2 = s2 / 512, h2 = (s2 % 512) / 128, g2 = s2 % 128;
      const float* B = (t2 == 0) ? bq1 : (t2 == 1) ? bkm1 : bvm1;
      bias[i] = B[h2 * 128 + g2];
    }
  }
}

// W2stack swizzled per group: 12 x (K=128, N=64, nBlk=1); group = t*4+h
__global__ __launch_bounds__(256) void k_cvt_w2stack(const float* __restrict__ Wq2,
                                                     const float* __restrict__ Wkm2,
                                                     const float* __restrict__ Wvm2,
                                                     __bf16* __restrict__ dst) {
  long tot = 12L * 128 * 64;
  for (long i = (long)blockIdx.x * blockDim.x + threadIdx.x; i < tot;
       i += (long)gridDim.x * blockDim.x) {
    int gi = (int)(i >> 13);
    long gidx = i & 8191;
    int k, e; bswz(gidx, 1, k, e);
    int t = gi / 4, h = gi % 4;
    const float* W = (t == 0) ? Wq2 : (t == 1) ? Wkm2 : Wvm2;
    dst[i] = (__bf16)W[h * 8192 + e * 128 + k];  // [H][E][HE]
  }
}

// went swizzled for attn B frags: [mat(16)][nt(4)][l16(16)][k2(16)]
//   mat = (kv*2+ea)*4 + h ; e = nt*16 + l16 ; d = k2 ; src W[h][e][d]
__global__ __launch_bounds__(256) void k_cvt_went(const float* __restrict__ Wke,
                                                  const float* __restrict__ Wka,
                                                  const float* __restrict__ Wve,
                                                  const float* __restrict__ Wva,
                                                  __bf16* __restrict__ dst) {
  long tot = 16L * 4 * 16 * 16;
  for (long i = (long)blockIdx.x * blockDim.x + threadIdx.x; i < tot;
       i += (long)gridDim.x * blockDim.x) {
    int k2 = (int)(i & 15);
    int l16 = (int)((i >> 4) & 15);
    int nt = (int)((i >> 8) & 3);
    int mat = (int)(i >> 10);
    int h = mat & 3, ea = (mat >> 2) & 1, kv = mat >> 3;
    int e = nt * 16 + l16, d = k2;
    const float* W = (kv == 0) ? ((ea == 0) ? Wke : Wka) : ((ea == 0) ? Wve : Wva);
    dst[i] = (__bf16)W[h * 1024 + e * 16 + d];
  }
}

// Wf^T swizzled: K=8192, N=256 (nBlk=4)
__global__ __launch_bounds__(256) void k_cvt_wf(const float* __restrict__ Wf,
                                                __bf16* __restrict__ dst) {
  long tot = 8192L * 256;
  for (long i = (long)blockIdx.x * blockDim.x + threadIdx.x; i < tot;
       i += (long)gridDim.x * blockDim.x) {
    int k, r; bswz(i, 4, k, r);
    dst[i] = (__bf16)Wf[(long)r * 8192 + k];
  }
}

// =============================================================
// generic bf16 WMMA GEMM, async-LDS staged, double buffered
// block 128x64, 8 waves; B pre-swizzled [K/32][N/64][2048]
// =============================================================
__global__ __launch_bounds__(256) void gemm_bf16(
    const __bf16* __restrict__ A, int lda, long aGS,
    const __bf16* __restrict__ Bsw, long bGS,
    float* __restrict__ Cf, int ldc, long cGS,
    __bf16* __restrict__ Cbf,
    const float* __restrict__ bias, int K, int relu) {
  __shared__ __bf16 As[2][128 * 32];
  __shared__ __bf16 Bs[2][2048];
  const int tid = threadIdx.x, lane = tid & 31, w = tid >> 5;
  const long g = blockIdx.z;
  A += g * aGS;
  Bsw += g * bGS;
  const long m0b = (long)blockIdx.x * 128;
  const int n0b = blockIdx.y * 64;
  const int nBlkCnt = gridDim.y;

  v8f acc[4];
  for (int j = 0; j < 4; ++j)
    for (int r = 0; r < 8; ++r) acc[j][r] = 0.f;

  auto issue_tile = [&](int buf, int kt) {
    for (int i = 0; i < 2; ++i) {           // A: 512 chunks of 16B
      int c = i * 256 + tid;
      int row = c >> 2, seg = c & 3;
      async_copy16(&As[buf][row * 32 + seg * 8],
                   A + (m0b + row) * (long)lda + kt + seg * 8);
    }
    {                                        // B: 256 chunks of 16B (linear)
      long boff = ((long)(kt >> 5) * nBlkCnt + blockIdx.y) * 2048;
      async_copy16(&Bs[buf][tid * 8], Bsw + boff + tid * 8);
    }
  };

  issue_tile(0, 0);
  const int nk = K >> 5;
  for (int it = 0; it < nk; ++it) {
    int buf = it & 1;
    if (it + 1 < nk) {
      issue_tile(buf ^ 1, (it + 1) << 5);
      async_wait_le3();   // 3 newest = next tile; current tile done (in-order)
    } else {
      async_wait_le0();
    }
    __syncthreads();

    v16bf af;
    {
      int Mo = (w * 16 + (lane & 15)) * 32 + ((lane < 16) ? 0 : 16);
      v8bf lo = *(const v8bf*)&As[buf][Mo];
      v8bf hi = *(const v8bf*)&As[buf][Mo + 8];
      af = cat16(lo, hi);
    }
    for (int jj = 0; jj < 4; ++jj) {
      int bo = (jj * 32 + lane) * 16;
      v8bf lo = *(const v8bf*)&Bs[buf][bo];
      v8bf hi = *(const v8bf*)&Bs[buf][bo + 8];
      acc[jj] = __builtin_amdgcn_wmma_f32_16x16x32_bf16(
          false, af, false, cat16(lo, hi), (short)0, acc[jj], false, false);
    }
    __syncthreads();
  }

  for (int jj = 0; jj < 4; ++jj) {
    int colL = n0b + jj * 16 + (lane & 15);
    long rowB = m0b + w * 16 + ((lane < 16) ? 0 : 8);
    for (int r = 0; r < 8; ++r) {
      float v = acc[jj][r];
      if (bias) v += bias[colL];
      if (relu) v = v > 0.f ? v : 0.f;
      long off = (rowB + r) * (long)ldc + colL;
      if (Cf) (Cf + g * cGS)[off] = v;
      if (Cbf) (Cbf + g * cGS)[off] = (__bf16)v;
    }
  }
}

// =============================================================
// w_head / b_term epilogue
// =============================================================
__global__ __launch_bounds__(256) void k_epi_wb(const float* __restrict__ hidden,
                                                const float* __restrict__ Wb2,
                                                const float* __restrict__ bb2,
                                                const float* __restrict__ Wh2,
                                                const float* __restrict__ bh2,
                                                float* __restrict__ wb) {
  int b = blockIdx.x * blockDim.x + threadIdx.x;
  if (b >= BSZ) return;
  const float* hr = hidden + (long)b * 192;
  float bt = bb2[0];
  for (int j = 0; j < 64; ++j) bt += hr[j] * Wb2[j];
  for (int hh = 0; hh < 4; ++hh) {
    float a = bh2[hh];
    for (int j = 0; j < 128; ++j) a += hr[64 + j] * Wh2[hh * 128 + j];
    wb[b * 8 + hh] = fabsf(a);
  }
  wb[b * 8 + 4] = bt;
}

// =============================================================
// fused per-row attention (128 threads = 4 waves, 1 block / row)
// =============================================================
__global__ __launch_bounds__(128) void k_attn(const float* __restrict__ obs,
                                              const float* __restrict__ qkv,   // [B][768]
                                              const float* __restrict__ wb,    // [B][8]
                                              const __bf16* __restrict__ wentG,// swizzled 16384
                                              __bf16* __restrict__ mixed) {    // [B][8192]
  __shared__ __bf16 entL[128 * 16];      // entities (row 127 = pad)
  __shared__ __bf16 wentS[16384];        // swizzled B frags
  __shared__ float qkvL[768];
  __shared__ float kvbuf[128 * 68];
  __shared__ float mixedL[8192];
  __shared__ float larr[128];
  __shared__ float parr[128];

  const int tid = threadIdx.x, lane = tid & 31, w = tid >> 5;
  const long b = blockIdx.x;
  const float* orow = obs + b * (long)INF;

  // async stage: swizzled weights (32KB) + qkv row (3KB)
  for (int c = tid; c < 2048; c += 128)
    async_copy16(&wentS[c * 8], wentG + c * 8);
  for (int c = tid; c < 192; c += 128)
    async_copy16(&qkvL[c * 4], qkv + b * 768 + c * 4);

  // entities: f32 -> bf16 (manual, needs convert)
  for (int i = tid; i < 2048; i += 128) {
    int row = i >> 4, col = i & 15;
    float v = 0.f;
    if (row < 64) v = orow[EN_START + row * 16 + col];
    else if (row < 127) v = orow[AL_START + (row - 64) * 16 + col];
    entL[i] = (__bf16)v;
  }
  for (int i = tid; i < 8192; i += 128) mixedL[i] = 0.f;

  float whead[4], bterm;
  {
    const float* p = wb + b * 8;
    whead[0] = p[0]; whead[1] = p[1]; whead[2] = p[2]; whead[3] = p[3];
    bterm = p[4];
  }
  async_wait_le0();
  __syncthreads();

  float u_reg = 0.f;
  for (int h = 0; h < NHEAD; ++h) {
    for (int kv = 0; kv < 2; ++kv) {
      __syncthreads();  // kvbuf ownership handoff
      for (int t2 = 0; t2 < 2; ++t2) {
        int mt = w + t2 * 4;                 // wave's M tiles {w, w+4}
        int ea = (mt < 4) ? 0 : 1;
        int mat = (kv * 2 + ea) * 4 + h;
        v16bf af;
        {
          int Mo = (mt * 16 + (lane & 15)) * 16;
          v8bf lo = *(const v8bf*)&entL[Mo];
          v8bf hi = *(const v8bf*)&entL[Mo + 8];
          v8bf zz;
          for (int j = 0; j < 8; ++j) zz[j] = (__bf16)0.0f;
          if (lane < 16) af = cat16(lo, hi);
          else af = cat16(zz, zz);           // K 16..31 pad
        }
        for (int nt = 0; nt < 4; ++nt) {
          v16bf bfr;
          if (lane < 16) {
            int bo = ((mat * 4 + nt) * 16 + lane) * 16;
            v8bf lo = *(const v8bf*)&wentS[bo];
            v8bf hi = *(const v8bf*)&wentS[bo + 8];
            bfr = cat16(lo, hi);
          } else {
            for (int j = 0; j < 16; ++j) bfr[j] = (__bf16)0.0f;  // K pad rows
          }
          v8f acc;
          for (int r = 0; r < 8; ++r) acc[r] = 0.f;
          acc = __builtin_amdgcn_wmma_f32_16x16x32_bf16(
              false, af, false, bfr, (short)0, acc, false, false);
          int colL = nt * 16 + (lane & 15);
          int rowB = mt * 16 + ((lane < 16) ? 0 : 8);
          for (int r = 0; r < 8; ++r) kvbuf[(rowB + r) * 68 + colL] = acc[r];
        }
      }
      __syncthreads();

      if (kv == 0) {
        const float* qh = &qkvL[h * 64];
        float dot = 0.f;
        if (tid == 0) {
          const float* kmh = &qkvL[256 + h * 64];
          for (int e = 0; e < 64; ++e) dot += qh[e] * kmh[e];
        } else {
          const float* kr = &kvbuf[(tid - 1) * 68];
          for (int e = 0; e < 64; ++e) dot += qh[e] * kr[e];
        }
        larr[tid] = dot * 0.125f;  // 1/sqrt(64)
        __syncthreads();
        float mx = -1e30f;
        for (int j = 0; j < 128; ++j) mx = fmaxf(mx, larr[j]);
        float p = __expf(larr[tid] - mx);
        parr[tid] = p;
        __syncthreads();
        float s = 0.f;
        for (int j = 0; j < 128; ++j) s += parr[j];
        u_reg = whead[h] * p / s;
      } else {
        const float* vr = (tid == 0) ? &qkvL[512 + h * 64] : &kvbuf[(tid - 1) * 68];
        for (int e = 0; e < 64; ++e) mixedL[e * 128 + tid] += u_reg * vr[e];
      }
    }
  }
  __syncthreads();
  for (int i = tid; i < 8192; i += 128)
    mixed[b * (long)KMIX + i] = (__bf16)(mixedL[i] + bterm);
}

// =============================================================
// launch
// =============================================================
static inline char* wsAlloc(char*& p, size_t bytes) {
  char* r = p;
  p += (bytes + 255) & ~(size_t)255;
  return r;
}

extern "C" void kernel_launch(void* const* d_in, const int* in_sizes, int n_in,
                              void* d_out, int out_size, void* d_ws, size_t ws_size,
                              hipStream_t stream) {
  const float* obs  = (const float*)d_in[0];
  const float* Wq1  = (const float*)d_in[1];
  const float* bq1  = (const float*)d_in[2];
  const float* Wq2  = (const float*)d_in[3];
  const float* Wkm1 = (const float*)d_in[4];
  const float* bkm1 = (const float*)d_in[5];
  const float* Wkm2 = (const float*)d_in[6];
  const float* Wke  = (const float*)d_in[7];
  const float* Wka  = (const float*)d_in[8];
  const float* Wvm1 = (const float*)d_in[9];
  const float* bvm1 = (const float*)d_in[10];
  const float* Wvm2 = (const float*)d_in[11];
  const float* Wve  = (const float*)d_in[12];
  const float* Wva  = (const float*)d_in[13];
  const float* Wb1  = (const float*)d_in[14];
  const float* bb1  = (const float*)d_in[15];
  const float* Wb2  = (const float*)d_in[16];
  const float* bb2  = (const float*)d_in[17];
  const float* Wh1  = (const float*)d_in[18];
  const float* bh1  = (const float*)d_in[19];
  const float* Wh2  = (const float*)d_in[20];
  const float* bh2  = (const float*)d_in[21];
  const float* Wf   = (const float*)d_in[22];
  const float* bfv  = (const float*)d_in[23];
  float* out = (float*)d_out;

  char* p = (char*)d_ws;
  __bf16* obs_bf   = (__bf16*)wsAlloc(p, (size_t)BSZ * INP * 2);
  __bf16* m_bf     = (__bf16*)wsAlloc(p, (size_t)BSZ * MFP * 2);
  __bf16* w1bigT   = (__bf16*)wsAlloc(p, (size_t)INP * 192 * 2);
  float*  bias192  = (float*)wsAlloc(p, 192 * 4);
  __bf16* w1stackT = (__bf16*)wsAlloc(p, (size_t)MFP * 1536 * 2);
  float*  bias1536 = (float*)wsAlloc(p, 1536 * 4);
  __bf16* w2stack  = (__bf16*)wsAlloc(p, 12L * 128 * 64 * 2);
  __bf16* wentB    = (__bf16*)wsAlloc(p, 16384 * 2);
  __bf16* wfT      = (__bf16*)wsAlloc(p, 8192L * 256 * 2);
  float*  hidden192= (float*)wsAlloc(p, (size_t)BSZ * 192 * 4);
  float*  wbArr    = (float*)wsAlloc(p, (size_t)BSZ * 8 * 4);
  __bf16* hid1536  = (__bf16*)wsAlloc(p, (size_t)BSZ * 1536 * 2);
  float*  qkv768   = (float*)wsAlloc(p, (size_t)BSZ * 768 * 4);
  __bf16* mixed    = (__bf16*)wsAlloc(p, (size_t)BSZ * KMIX * 2);

  // --- conversions / swizzles ---
  k_cvt_obs<<<dim3(4096), dim3(256), 0, stream>>>(obs, obs_bf);
  k_cvt_m<<<dim3(512), dim3(256), 0, stream>>>(obs, m_bf);
  k_cvt_w1big<<<dim3(1024), dim3(256), 0, stream>>>(Wb1, Wh1, bb1, bh1, w1bigT, bias192);
  k_cvt_w1stack<<<dim3(192), dim3(256), 0, stream>>>(Wq1, Wkm1, Wvm1, bq1, bkm1, bvm1,
                                                     w1stackT, bias1536);
  k_cvt_w2stack<<<dim3(384), dim3(256), 0, stream>>>(Wq2, Wkm2, Wvm2, w2stack);
  k_cvt_went<<<dim3(64), dim3(256), 0, stream>>>(Wke, Wka, Wve, Wva, wentB);
  k_cvt_wf<<<dim3(2048), dim3(256), 0, stream>>>(Wf, wfT);

  // --- hyper-net hidden: [B,2080] x [2080,192] (+bias, relu) -> f32 ---
  gemm_bf16<<<dim3(32, 3, 1), dim3(256), 0, stream>>>(
      obs_bf, INP, 0, w1bigT, 0, hidden192, 192, 0, (__bf16*)nullptr,
      bias192, INP, 1);
  k_epi_wb<<<dim3(16), dim3(256), 0, stream>>>(hidden192, Wb2, bb2, Wh2, bh2, wbArr);

  // --- headed MLP layer 1: [B,32] x [32,1536] (+bias, relu) -> bf16 ---
  gemm_bf16<<<dim3(32, 24, 1), dim3(256), 0, stream>>>(
      m_bf, MFP, 0, w1stackT, 0, (float*)nullptr, 1536, 0, hid1536,
      bias1536, MFP, 1);

  // --- headed MLP layer 2 (grouped/block-diag): 12 x ([B,128] x [128,64]) ---
  gemm_bf16<<<dim3(32, 1, 12), dim3(256), 0, stream>>>(
      hid1536, 1536, 128, w2stack, 8192, qkv768, 768, 64, (__bf16*)nullptr,
      (const float*)nullptr, 128, 0);

  // --- fused attention -> mixed (bf16) ---
  k_attn<<<dim3(BSZ), dim3(128), 0, stream>>>(obs, qkv768, wbArr, wentB, mixed);

  // --- final GEMM: [B,8192] x [8192,256] (+bf) -> out f32 ---
  gemm_bf16<<<dim3(32, 4, 1), dim3(256), 0, stream>>>(
      mixed, KMIX, 0, wfT, 0, out, RNN, 0, (__bf16*)nullptr, bfv, KMIX, 0);
}
